// GraphQueryAttention_64484638982858
// MI455X (gfx1250) — matrix operand order repaired and checked by reading, not compiled, not run on hardware
//
#include <hip/hip_runtime.h>

// ---------------- problem constants ----------------
#define Bq   2
#define Sq   2048
#define HIDq 2048
#define Hq   16
#define KVHq 4
#define HDq  128
#define NREP 4
#define Mtok (Bq*Sq)           // 4096
#define SCALEF 0.08838834764831843f   // 1/sqrt(128)

typedef unsigned int u32;
typedef unsigned short u16;
typedef __attribute__((ext_vector_type(16))) __bf16 v16bf;
typedef __attribute__((ext_vector_type(8)))  float  v8f;
typedef __attribute__((ext_vector_type(4)))  u32    v4u;
typedef __attribute__((ext_vector_type(8)))  int    v8i;
typedef __attribute__((ext_vector_type(4)))  int    v4i;

union Frag { v16bf v; u32 u[8]; };

__device__ __forceinline__ u16 f2bf(float f) {
  u32 x = __float_as_uint(f);
  u32 r = (x + 0x7fffu + ((x >> 16) & 1u)) >> 16;
  return (u16)r;
}

__device__ __forceinline__ v8f wmma_bf16(const Frag& a, const Frag& b, v8f c) {
  return __builtin_amdgcn_wmma_f32_16x16x32_bf16(false, a.v, false, b.v,
                                                 (short)0, c, false, false);
}

// ------- Tensor Data Mover: 2D tile (dword units) global -> LDS -------
// D# per CDNA5 ISA ch.8: group0 = {flags, lds_addr, global_addr, type=2},
// group1 = {mask/data_size, tensor dims, tile dims, dim0 stride}.
// This toolchain's builtin takes 6 args (extra v8i group before cpol).
__device__ __forceinline__ void tdm_load_2d(unsigned lds_off, const void* gptr,
                                            unsigned width_dw, unsigned height,
                                            unsigned stride_dw) {
  unsigned long long ga = (unsigned long long)gptr;
  v4u g0 = { 1u,                                   // count=1, user descriptor
             lds_off,                               // LDS byte address
             (u32)ga,                               // global addr [31:0]
             (u32)((ga >> 32) & 0x1ffffffu) | (2u << 30) }; // addr[56:32] | type=2
  v8i g1;
  g1[0] = (int)(2u << 16);                               // data_size = 4 bytes
  g1[1] = (int)((width_dw & 0xffffu) << 16);             // tensor_dim0 lo16
  g1[2] = (int)(((width_dw >> 16) & 0xffffu) | ((height & 0xffffu) << 16)); // dim0 hi | dim1 lo
  g1[3] = (int)(((height >> 16) & 0xffffu) | ((width_dw & 0xffffu) << 16)); // dim1 hi | tile_dim0
  g1[4] = (int)(height & 0xffffu);                       // tile_dim1 | tile_dim2=0
  g1[5] = (int)stride_dw;                                // tensor_dim0_stride lo32
  g1[6] = 0;
  g1[7] = 0;
  v4i z4 = { 0, 0, 0, 0 };
  v8i z8 = { 0, 0, 0, 0, 0, 0, 0, 0 };
  __builtin_amdgcn_tensor_load_to_lds(g0, g1, z4, z4, z8, 0);
}

// ---------------- f32 -> bf16 convert ----------------
__global__ void k_cvt(const float* __restrict__ in, u16* __restrict__ out, int n) {
  int i = blockIdx.x * blockDim.x + threadIdx.x;
  if (i < n) out[i] = f2bf(in[i]);
}

// -------- transpose + convert: W (K x N) f32 -> Wt (N x K) bf16 --------
__global__ void k_tcvt(const float* __restrict__ W, u16* __restrict__ Wt, int K, int N) {
  __shared__ float t[32][33];
  int n0 = blockIdx.x * 32, k0 = blockIdx.y * 32;
  int tx = threadIdx.x, ty = threadIdx.y;          // block (32,8)
  for (int i = 0; i < 4; ++i)
    t[ty + i * 8][tx] = W[(size_t)(k0 + ty + i * 8) * N + n0 + tx];
  __syncthreads();
  for (int i = 0; i < 4; ++i)
    Wt[(size_t)(n0 + ty + i * 8) * K + k0 + tx] = f2bf(t[tx][ty + i * 8]);
}

// ---------------- bf16 tiled GEMM: C = A(MxK) * Bt(NxK)^T ----------------
// 128x128 tile / WG, 8 waves (2x4), wave -> 64x32 via 4x2 WMMA accums.
// Cooperative copy fully unrolled: each thread moves 1 uint4 of A and B in
// rows [0,64) and 1 each in rows [64,128); global pointers advance by a
// constant 64B per k-step (no per-iteration address multiplies).
// out_mode: 0 = bf16 row-major, 1 = f32 row-major, 2 = bf16 scattered as V^T
__global__ __launch_bounds__(256, 1)
void k_gemm(const u16* __restrict__ A, const u16* __restrict__ Bt, void* Cv,
            int Ndim, int Kdim, int out_mode) {
  __shared__ __align__(16) u16 lA[128 * 32];
  __shared__ __align__(16) u16 lB[128 * 32];      // [n][k]
  const int tid = threadIdx.x, wave = tid >> 5, lane = tid & 31;
  const int lh = lane >> 4, ln = lane & 15;
  const int m0 = blockIdx.y * 128, n0 = blockIdx.x * 128;
  const int wm = wave >> 2, wn = wave & 3;

  // copy assignments (row = tid>>2 and +64, quarter = tid&3)
  const int crow = tid >> 2, cq = tid & 3;
  const u16* pA0 = A + (size_t)(m0 + crow) * Kdim + cq * 8;
  const u16* pA1 = A + (size_t)(m0 + crow + 64) * Kdim + cq * 8;
  const u16* pB0 = Bt + (size_t)(n0 + crow) * Kdim + cq * 8;
  const u16* pB1 = Bt + (size_t)(n0 + crow + 64) * Kdim + cq * 8;
  u16* sA0 = &lA[crow * 32 + cq * 8];
  u16* sA1 = &lA[(crow + 64) * 32 + cq * 8];
  u16* sB0 = &lB[crow * 32 + cq * 8];
  u16* sB1 = &lB[(crow + 64) * 32 + cq * 8];

  v8f acc[4][2];
  for (int mt = 0; mt < 4; ++mt)
    for (int nt = 0; nt < 2; ++nt)
      for (int r = 0; r < 8; ++r) acc[mt][nt][r] = 0.f;

  for (int k0 = 0; k0 < Kdim; k0 += 32) {
    __syncthreads();
    *(uint4*)sA0 = *(const uint4*)pA0;
    *(uint4*)sA1 = *(const uint4*)pA1;
    *(uint4*)sB0 = *(const uint4*)pB0;
    *(uint4*)sB1 = *(const uint4*)pB1;
    pA0 += 32; pA1 += 32; pB0 += 32; pB1 += 32;
    __syncthreads();

    Frag af[4];
    for (int mt = 0; mt < 4; ++mt) {
      const u16* ap = &lA[(wm * 64 + mt * 16 + ln) * 32 + lh * 8];
      for (int i = 0; i < 4; ++i) {
        af[mt].u[i]     = *(const u32*)(ap + 2 * i);
        af[mt].u[4 + i] = *(const u32*)(ap + 16 + 2 * i);
      }
    }
    for (int nt = 0; nt < 2; ++nt) {
      Frag bf;
      const u16* bp = &lB[(wn * 32 + nt * 16 + ln) * 32 + lh * 8];
      for (int v = 0; v < 8; ++v) bf.u[v] = *(const u32*)(bp + 2 * v + (v & 4) * 2);
      for (int mt = 0; mt < 4; ++mt)
        acc[mt][nt] = wmma_bf16(af[mt], bf, acc[mt][nt]);
    }
  }

  for (int mt = 0; mt < 4; ++mt)
    for (int nt = 0; nt < 2; ++nt)
      for (int r = 0; r < 8; ++r) {
        int m = m0 + wm * 64 + mt * 16 + r + lh * 8;
        int n = n0 + wn * 32 + nt * 16 + ln;
        float v = acc[mt][nt][r];
        if (out_mode == 1) {
          ((float*)Cv)[(size_t)m * Ndim + n] = v;
        } else if (out_mode == 0) {
          ((u16*)Cv)[(size_t)m * Ndim + n] = f2bf(v);
        } else {      // scatter as V^T: (B,KVH,HD,S)
          int bb = m >> 11, ss = m & (Sq - 1);
          int kv = n >> 7,  hd = n & (HDq - 1);
          ((u16*)Cv)[(((size_t)(bb * KVHq + kv) * HDq) + hd) * Sq + ss] = f2bf(v);
        }
      }
}

// ---------------- flash attention (bf16 WMMA, f32 accumulate) ----------------
// Grid: (S/128, H, B); block 256 (8 waves, 16 q-rows each); 64-key tiles.
__global__ __launch_bounds__(256, 1)
void k_attn(const u16* __restrict__ Qb, const u16* __restrict__ Kb,
            const u16* __restrict__ Vtb, const float* __restrict__ bias,
            const int* __restrict__ mask, u16* __restrict__ Ob) {
  __shared__ __align__(16) u16 lK[64 * 128];   // [key][hd]
  __shared__ __align__(16) u16 lV[128 * 64];   // [hd][key]
  __shared__ __align__(16) u16 lP[8][16 * 64]; // per-wave P scratch [row][key]

  const int qt = blockIdx.x, h = blockIdx.y, b = blockIdx.z;
  const int kvh = h / NREP;
  const int tid = threadIdx.x, wave = tid >> 5, lane = tid & 31;
  const int lh = lane >> 4, ln = lane & 15;
  const int q0 = qt * 128 + wave * 16;

  // Q A-fragments cached in VGPRs (reused for all key tiles)
  Frag qf[4];
  {
    const u16* qrow = Qb + (size_t)(b * Sq + q0 + ln) * (Hq * HDq) + h * HDq;
    for (int c = 0; c < 4; ++c)
      for (int i = 0; i < 4; ++i) {
        int ko = c * 32 + lh * 8 + 2 * i;
        qf[c].u[i]     = *(const u32*)(qrow + ko);
        qf[c].u[4 + i] = *(const u32*)(qrow + 16 + ko);
      }
  }

  v8f O[8];
  for (int ns = 0; ns < 8; ++ns)
    for (int r = 0; r < 8; ++r) O[ns][r] = 0.f;
  float mrow[8], lrow[8];
  for (int r = 0; r < 8; ++r) { mrow[r] = -3e38f; lrow[r] = 0.f; }

  for (int kt = 0; kt < Sq / 64; ++kt) {
    const int k0 = kt * 64;
    __syncthreads();
    if (wave == 0) {   // one wave drives the Tensor Data Mover for both tiles
      tdm_load_2d((unsigned)(unsigned long long)(void*)lK,
                  Kb + ((size_t)(b * Sq + k0) * KVHq + kvh) * HDq,
                  /*w dwords*/ 64, /*rows*/ 64, /*stride dwords*/ (KVHq * HDq) / 2);
      tdm_load_2d((unsigned)(unsigned long long)(void*)lV,
                  Vtb + ((size_t)(b * KVHq + kvh) * HDq) * Sq + k0,
                  /*w dwords*/ 32, /*rows*/ 128, /*stride dwords*/ Sq / 2);
      __builtin_amdgcn_s_wait_tensorcnt(0);
    }
    __syncthreads();

    // ---- scores: 16 x 64, WMMA over HD=128 ----
    v8f Sc[4];
    float newm[8];
    for (int r = 0; r < 8; ++r) newm[r] = mrow[r];
    for (int nt = 0; nt < 4; ++nt) {
      v8f a;
      for (int r = 0; r < 8; ++r) a[r] = 0.f;
      for (int c = 0; c < 4; ++c) {
        Frag bf;
        const u16* kp = &lK[(nt * 16 + ln) * 128 + c * 32 + lh * 8];
        for (int v = 0; v < 8; ++v) bf.u[v] = *(const u32*)(kp + 2 * v + (v & 4) * 2);
        a = wmma_bf16(qf[c], bf, a);
      }
      int km = mask[b * Sq + k0 + nt * 16 + ln];
      const float* bp = bias + ((size_t)h * Sq + (q0 + lh * 8)) * Sq + k0 + nt * 16 + ln;
      for (int r = 0; r < 8; ++r) {
        float v = a[r] * SCALEF + bp[(size_t)r * Sq];
        if (km == 0) v = -1e30f;
        a[r] = v;
        newm[r] = fmaxf(newm[r], v);
      }
      Sc[nt] = a;
    }
    // row max across the 16 lanes of each half
    for (int r = 0; r < 8; ++r) {
      float v = newm[r];
      v = fmaxf(v, __shfl_xor(v, 1, 32));
      v = fmaxf(v, __shfl_xor(v, 2, 32));
      v = fmaxf(v, __shfl_xor(v, 4, 32));
      v = fmaxf(v, __shfl_xor(v, 8, 32));
      newm[r] = v;
    }
    // online-softmax rescale
    for (int r = 0; r < 8; ++r) {
      float rs = __expf(mrow[r] - newm[r]);
      mrow[r] = newm[r];
      lrow[r] *= rs;
      for (int ns = 0; ns < 8; ++ns) O[ns][r] *= rs;
    }
    // P = exp(S - m), row sums
    for (int nt = 0; nt < 4; ++nt)
      for (int r = 0; r < 8; ++r) Sc[nt][r] = __expf(Sc[nt][r] - newm[r]);
    for (int r = 0; r < 8; ++r) {
      float s = Sc[0][r] + Sc[1][r] + Sc[2][r] + Sc[3][r];
      s += __shfl_xor(s, 1, 32);
      s += __shfl_xor(s, 2, 32);
      s += __shfl_xor(s, 4, 32);
      s += __shfl_xor(s, 8, 32);
      lrow[r] += s;
    }
    // C-layout -> A-layout via per-wave LDS scratch
    u16* pw = &lP[wave][0];
    for (int nt = 0; nt < 4; ++nt)
      for (int r = 0; r < 8; ++r)
        pw[(r + lh * 8) * 64 + nt * 16 + ln] = f2bf(Sc[nt][r]);
    asm volatile("s_wait_dscnt 0" ::: "memory");   // wave-local RAW on lP

    // ---- O += P * V ----
    for (int c = 0; c < 2; ++c) {
      Frag pf;
      const u16* pr = pw + ln * 64 + c * 32 + lh * 8;
      for (int i = 0; i < 4; ++i) {
        pf.u[i]     = *(const u32*)(pr + 2 * i);
        pf.u[4 + i] = *(const u32*)(pr + 16 + 2 * i);
      }
      for (int ns = 0; ns < 8; ++ns) {
        Frag vf;
        const u16* vp = &lV[(ns * 16 + ln) * 64 + c * 32 + lh * 8];
        for (int v = 0; v < 8; ++v) vf.u[v] = *(const u32*)(vp + 2 * v + (v & 4) * 2);
        O[ns] = wmma_bf16(pf, vf, O[ns]);
      }
    }
  }

  // epilogue: normalize, store head_out bf16 (B, S, H*HD)
  for (int ns = 0; ns < 8; ++ns)
    for (int r = 0; r < 8; ++r) {
      float o = O[ns][r] / lrow[r];
      Ob[(size_t)(b * Sq + q0 + r + lh * 8) * (Hq * HDq) + h * HDq + ns * 16 + ln] = f2bf(o);
    }
}

// ---------------- host-side orchestration ----------------
extern "C" void kernel_launch(void* const* d_in, const int* in_sizes, int n_in,
                              void* d_out, int out_size, void* d_ws, size_t ws_size,
                              hipStream_t stream) {
  const float* Xf   = (const float*)d_in[0];  // (B,S,HID)
  const int*   msk  = (const int*)d_in[1];    // (B,S)
  const float* bias = (const float*)d_in[2];  // (H,S,S)
  const float* Wq   = (const float*)d_in[3];  // (2048,2048)
  const float* Wk   = (const float*)d_in[4];  // (2048,512)
  const float* Wv   = (const float*)d_in[5];  // (2048,512)
  const float* Wo   = (const float*)d_in[6];  // (2048,2048)
  float* out = (float*)d_out;

  char* ws = (char*)d_ws;
  const size_t MB = 1024ull * 1024ull;
  u16* Xb  = (u16*)(ws + 0 * MB);    // 16 MB  (4096 x 2048)
  u16* Wqt = (u16*)(ws + 16 * MB);   //  8 MB  (2048 x 2048)
  u16* Wkt = (u16*)(ws + 24 * MB);   //  2 MB  (512  x 2048)
  u16* Wvt = (u16*)(ws + 26 * MB);   //  2 MB  (512  x 2048)
  u16* Wot = (u16*)(ws + 28 * MB);   //  8 MB  (2048 x 2048)
  u16* Qb  = (u16*)(ws + 36 * MB);   // 16 MB  (4096 x 2048)
  u16* Kb  = (u16*)(ws + 52 * MB);   //  4 MB  (4096 x 512)
  u16* Vtb = (u16*)(ws + 56 * MB);   //  4 MB  (B,KVH,HD,S)
  u16* Ob  = (u16*)(ws + 60 * MB);   // 16 MB  (4096 x 2048)

  // 1) convert activations + transpose-convert weights to bf16
  {
    int n = Mtok * HIDq;
    k_cvt<<<(n + 255) / 256, 256, 0, stream>>>(Xf, Xb, n);
  }
  k_tcvt<<<dim3(HIDq / 32, HIDq / 32), dim3(32, 8), 0, stream>>>(Wq, Wqt, HIDq, HIDq);
  k_tcvt<<<dim3((KVHq * HDq) / 32, HIDq / 32), dim3(32, 8), 0, stream>>>(Wk, Wkt, HIDq, KVHq * HDq);
  k_tcvt<<<dim3((KVHq * HDq) / 32, HIDq / 32), dim3(32, 8), 0, stream>>>(Wv, Wvt, HIDq, KVHq * HDq);
  k_tcvt<<<dim3(HIDq / 32, HIDq / 32), dim3(32, 8), 0, stream>>>(Wo, Wot, HIDq, HIDq);

  // 2) projections (bf16 WMMA GEMMs)
  k_gemm<<<dim3(HIDq / 128, Mtok / 128), 256, 0, stream>>>(Xb, Wqt, Qb, HIDq, HIDq, 0);
  k_gemm<<<dim3((KVHq * HDq) / 128, Mtok / 128), 256, 0, stream>>>(Xb, Wkt, Kb, KVHq * HDq, HIDq, 0);
  k_gemm<<<dim3((KVHq * HDq) / 128, Mtok / 128), 256, 0, stream>>>(Xb, Wvt, Vtb, KVHq * HDq, HIDq, 2);

  // 3) flash attention with TDM-staged K/V tiles
  k_attn<<<dim3(Sq / 128, Hq, Bq), 256, 0, stream>>>(Qb, Kb, Vtb, bias, msk, Ob);

  // 4) output projection -> f32 d_out
  k_gemm<<<dim3(HIDq / 128, Mtok / 128), 256, 0, stream>>>(Ob, Wot, out, HIDq, HIDq, 1);
}
